// ContractGNN_3539053052011
// MI455X (gfx1250) — matrix operand order repaired and checked by reading, not compiled, not run on hardware
//
#include <hip/hip_runtime.h>
#include <hip/hip_bf16.h>
#include <cstdint>

#define HCH 64
#define GNUM 128
#define NEG_SLOPE 0.2f

typedef __bf16 bf16_t;
typedef bf16_t v16bf  __attribute__((ext_vector_type(16)));
typedef bf16_t v4bf   __attribute__((ext_vector_type(4)));
typedef float  v8f    __attribute__((ext_vector_type(8)));
typedef uint32_t v4u32 __attribute__((ext_vector_type(4)));
typedef unsigned int v4u __attribute__((ext_vector_type(4)));
typedef int    v4i    __attribute__((ext_vector_type(4)));
typedef int    v8i    __attribute__((ext_vector_type(8)));

union Frag16 { v4u32 q[2]; uint32_t u[8]; v16bf v; };

// ---------------- atomic float max (sign-split trick, works with -inf init) ----
__device__ __forceinline__ void atomicMaxF(float* addr, float val) {
    if (val >= 0.0f) atomicMax((int*)addr, __float_as_int(val));
    else             atomicMin((unsigned int*)addr, __float_as_uint(val));
}

__device__ __forceinline__ float lrelu(float v) { return v > 0.0f ? v : NEG_SLOPE * v; }

__device__ __forceinline__ void edge_ep(int e, int E, const int* __restrict__ ei,
                                        int& s, int& d) {
    if (e < E) { s = ei[e]; d = ei[E + e]; }
    else       { s = e - E; d = s; }          // appended self loops
}

// ---------------- WMMA GEMM + fused attention-coefficient dots:
//   h[N,64] = X[N,K](f32) @ W[K,64]   (bf16 WMMA, f32 accumulate)
//   als[i] = h[i,:]·a_src, ald[i] = h[i,:]·a_dst  (fused epilogue)
// One 256-thread block = 8 waves, each wave produces a 16x64 tile.
template<int K>
__global__ __launch_bounds__(256)
void gat_gemm_wmma(const float* __restrict__ X, const bf16_t* __restrict__ WT,
                   const float* __restrict__ a_src, const float* __restrict__ a_dst,
                   float* __restrict__ Hout, float* __restrict__ als,
                   float* __restrict__ ald, int n) {
    constexpr int KP = K + 8;     // +4 dword row pad: 16B-aligned rows, conflict-free
    __shared__ bf16_t sX[128 * KP];
    __shared__ bf16_t sWT[64 * KP];

    const int tid  = threadIdx.x;
    const int wave = tid >> 5;
    const int lane = tid & 31;
    const int rowBase = blockIdx.x * 128;

    // ---- Stage X tile: float4 global loads -> packed bf16x4 LDS stores ----
    const float4* X4 = (const float4*)X;
    for (int idx = tid; idx < 128 * K / 4; idx += 256) {
        int r = (idx * 4) / K, c = (idx * 4) % K;
        int gr = rowBase + r;
        float4 f = make_float4(0.f, 0.f, 0.f, 0.f);
        if (gr < n) f = X4[(gr * K + c) >> 2];
        v4bf pk = { (bf16_t)f.x, (bf16_t)f.y, (bf16_t)f.z, (bf16_t)f.w };
        *(v4bf*)&sX[r * KP + c] = pk;
    }

    // ---- Stage W^T [64][K] via Tensor Data Mover; HW pad inserts 4 dwords
    //      after each K/2-dword row -> LDS row stride KP elements. ----
#if __has_builtin(__builtin_amdgcn_tensor_load_to_lds)
    if (wave == 0) {
        unsigned long long ga = (unsigned long long)(uintptr_t)WT;
        unsigned int lds = (unsigned int)(uintptr_t)&sWT[0];
        v4u g0 = { 1u,                                    // count=1
                   lds,                                   // lds_addr
                   (unsigned int)(ga & 0xffffffffu),      // global_addr[31:0]
                   (unsigned int)((ga >> 32) & 0x01ffffffu) | (2u << 30) }; // [56:32]|type=2
        const unsigned padc = (K == 128) ? 5u : 4u;       // pad every 64 / 32 dwords
        v8i g1;
        g1[0] = (int)((1u << 16)                          // data_size = 2 bytes
                    | (1u << 20)                          // pad_enable
                    | (padc << 22)                        // pad_interval
                    | (3u << 25));                        // pad_amount = 4 dwords
        g1[1] = (int)(((unsigned)K & 0xffffu) << 16);     // tensor_dim0 = K (lo16)
        g1[2] = (int)(64u << 16);                         // dim0 hi=0 | tensor_dim1 = 64
        g1[3] = (int)((unsigned)K << 16);                 // dim1 hi=0 | tile_dim0 = K
        g1[4] = 64;                                       // tile_dim1 = 64, tile_dim2 = 0
        g1[5] = K;                                        // tensor_dim0_stride = K
        g1[6] = 0; g1[7] = 0;
        v4i g2 = {0, 0, 0, 0}, g3 = {0, 0, 0, 0};
#if __clang_major__ >= 23
        v8i g4 = {0, 0, 0, 0, 0, 0, 0, 0};
        __builtin_amdgcn_tensor_load_to_lds(g0, g1, g2, g3, g4, 0);
#else
        __builtin_amdgcn_tensor_load_to_lds(g0, g1, g2, g3, 0);
#endif
        __builtin_amdgcn_s_wait_tensorcnt(0);
    }
#else
    for (int idx = tid; idx < 64 * K; idx += 256) {
        int r = idx / K, c = idx - r * K;
        sWT[r * KP + c] = WT[r * K + c];
    }
#endif
    __syncthreads();

    v8f acc[4];
    #pragma unroll
    for (int t = 0; t < 4; ++t)
        #pragma unroll
        for (int i = 0; i < 8; ++i) acc[t][i] = 0.0f;

    const int r0l   = wave * 16;
    const int arow  = lane & 15;
    const int akoff = (lane < 16) ? 0 : 8;    // ISA 16-bit A layout
    const int bn    = lane & 15;
    const int bkoff = (lane < 16) ? 0 : 8;    // B: same K layout as A
    const v4u32* sXrowQ = (const v4u32*)&sX[(r0l + arow) * KP];   // 16B-aligned

    for (int kb = 0; kb < K; kb += 32) {
        Frag16 a;
        a.q[0] = sXrowQ[(kb + akoff) >> 3];          // K = kb+akoff   .. +7
        a.q[1] = sXrowQ[(kb + 16 + akoff) >> 3];     // K = kb+16+akoff.. +7
        #pragma unroll
        for (int t = 0; t < 4; ++t) {
            const v4u32* sWrowQ = (const v4u32*)&sWT[(t * 16 + bn) * KP];
            Frag16 b;
            b.q[0] = sWrowQ[(kb + bkoff) >> 3];
            b.q[1] = sWrowQ[(kb + 16 + bkoff) >> 3];
            acc[t] = __builtin_amdgcn_wmma_f32_16x16x32_bf16(
                false, a.v, false, b.v, (short)0, acc[t], false, false);
        }
    }

    // C/D layout: VGPR v -> row v (lanes 0-15) / row v+8 (lanes 16-31), col = lane&15
    const int rofs = (lane >> 4) << 3;
    #pragma unroll
    for (int t = 0; t < 4; ++t) {
        int col = t * 16 + (lane & 15);
        #pragma unroll
        for (int v = 0; v < 8; ++v) {
            int gr = rowBase + r0l + rofs + v;
            if (gr < n) Hout[gr * HCH + col] = acc[t][v];
        }
    }

    // Fused alpha epilogue: per-row dot with a_src/a_dst + shfl-xor reduction
    float asv[4], adv[4];
    #pragma unroll
    for (int t = 0; t < 4; ++t) {
        asv[t] = a_src[t * 16 + (lane & 15)];
        adv[t] = a_dst[t * 16 + (lane & 15)];
    }
    #pragma unroll
    for (int v = 0; v < 8; ++v) {
        float s = 0.0f, d = 0.0f;
        #pragma unroll
        for (int t = 0; t < 4; ++t) { s += acc[t][v] * asv[t]; d += acc[t][v] * adv[t]; }
        #pragma unroll
        for (int msk = 1; msk <= 8; msk <<= 1) {   // stays within each 16-lane half
            s += __shfl_xor(s, msk, 32);
            d += __shfl_xor(d, msk, 32);
        }
        int gr = rowBase + r0l + rofs + v;
        if ((lane & 15) == 0 && gr < n) { als[gr] = s; ald[gr] = d; }
    }
}

// ---------------- small utility kernels ----------------
__global__ void fill_kernel(float* __restrict__ p, float v, int n) {
    int t = blockIdx.x * blockDim.x + threadIdx.x;
    if (t < n) p[t] = v;
}

// initialize aggregation buffer with broadcast bias (folds "+ b" into the scatter)
__global__ void fill_bias_kernel(float* __restrict__ p, const float* __restrict__ b, int n) {
    int t = blockIdx.x * blockDim.x + threadIdx.x;
    if (t < n * HCH) p[t] = b[t & (HCH - 1)];
}

__global__ void relu_kernel(float* __restrict__ p, int n) {
    int t = blockIdx.x * blockDim.x + threadIdx.x;
    if (t < n * HCH) p[t] = fmaxf(p[t], 0.0f);
}

__global__ void wt_convert_kernel(const float* __restrict__ W, bf16_t* __restrict__ WT, int K) {
    int t = blockIdx.x * blockDim.x + threadIdx.x;
    if (t >= K * HCH) return;
    int k = t / HCH, nn = t - k * HCH;        // W row-major [K][64]
    WT[nn * K + k] = (bf16_t)W[t];            // W^T [64][K]
}

__global__ void edge_max_kernel(const int* __restrict__ ei, int E, int ET,
                                const float* __restrict__ als, const float* __restrict__ ald,
                                float* __restrict__ m) {
    int e = blockIdx.x * blockDim.x + threadIdx.x;
    if (e >= ET) return;
    int s, d; edge_ep(e, E, ei, s, d);
    atomicMaxF(&m[d], lrelu(als[s] + ald[d]));
}

__global__ void edge_expsum_kernel(const int* __restrict__ ei, int E, int ET,
                                   const float* __restrict__ als, const float* __restrict__ ald,
                                   const float* __restrict__ m, float* __restrict__ denom) {
    int e = blockIdx.x * blockDim.x + threadIdx.x;
    if (e >= ET) return;
    int s, d; edge_ep(e, E, ei, s, d);
    atomicAdd(&denom[d], __expf(lrelu(als[s] + ald[d]) - m[d]));
}

// 4 threads per edge, each scatters 16 channels
__global__ void edge_aggr_kernel(const int* __restrict__ ei, int E, int ET,
                                 const float* __restrict__ als, const float* __restrict__ ald,
                                 const float* __restrict__ m, const float* __restrict__ denom,
                                 const float* __restrict__ h, float* __restrict__ out) {
    int t = blockIdx.x * blockDim.x + threadIdx.x;
    int e = t >> 2;
    if (e >= ET) return;
    int part = t & 3;
    int s, d; edge_ep(e, E, ei, s, d);
    float attn = __expf(lrelu(als[s] + ald[d]) - m[d]) / denom[d];
    const float4* hr = (const float4*)(h + s * HCH + part * 16);
    float* orow = out + d * HCH + part * 16;
    #pragma unroll
    for (int q = 0; q < 4; ++q) {
        float4 hv = hr[q];
        atomicAdd(&orow[q * 4 + 0], hv.x * attn);
        atomicAdd(&orow[q * 4 + 1], hv.y * attn);
        atomicAdd(&orow[q * 4 + 2], hv.z * attn);
        atomicAdd(&orow[q * 4 + 3], hv.w * attn);
    }
}

__global__ void pool_accum_kernel(const float* __restrict__ x, const int* __restrict__ batch,
                                  float* __restrict__ gmean, float* __restrict__ gmax,
                                  float* __restrict__ gcnt, int n) {
    int t = blockIdx.x * blockDim.x + threadIdx.x;
    if (t >= n * HCH) return;
    int node = t >> 6, c = t & 63;
    int g = batch[node];
    float v = x[t];
    atomicAdd(&gmean[g * HCH + c], v);
    atomicMaxF(&gmax[g * HCH + c], v);
    if (c == 0) atomicAdd(&gcnt[g], 1.0f);
}

__global__ void head_kernel(const float* __restrict__ gmean, const float* __restrict__ gmax,
                            const float* __restrict__ gcnt, const float* __restrict__ lw,
                            const float* __restrict__ lb, float* __restrict__ out, int C) {
    int t = blockIdx.x * blockDim.x + threadIdx.x;
    if (t >= GNUM * C) return;
    int g = t / C, c = t - g * C;
    float cnt = fmaxf(gcnt[g], 1.0f);
    float acc = lb[c];
    for (int k = 0; k < HCH; ++k) acc += (gmean[g * HCH + k] / cnt) * lw[k * C + c];
    for (int k = 0; k < HCH; ++k) acc += gmax[g * HCH + k] * lw[(HCH + k) * C + c];
    out[t] = acc;
}

// ---------------- launch ----------------
extern "C" void kernel_launch(void* const* d_in, const int* in_sizes, int n_in,
                              void* d_out, int out_size, void* d_ws, size_t ws_size,
                              hipStream_t stream) {
    const float* x     = (const float*)d_in[0];
    const int*   ei    = (const int*)d_in[1];
    const int*   batch = (const int*)d_in[2];
    const float* W[3]  = {(const float*)d_in[3], (const float*)d_in[7],  (const float*)d_in[11]};
    const float* As[3] = {(const float*)d_in[4], (const float*)d_in[8],  (const float*)d_in[12]};
    const float* Ad[3] = {(const float*)d_in[5], (const float*)d_in[9],  (const float*)d_in[13]};
    const float* B[3]  = {(const float*)d_in[6], (const float*)d_in[10], (const float*)d_in[14]};
    const float* lw = (const float*)d_in[15];
    const float* lb = (const float*)d_in[16];
    float* out = (float*)d_out;

    const int N  = in_sizes[2];
    const int E  = in_sizes[1] / 2;
    const int ET = E + N;
    const int C  = in_sizes[16];

    // carve workspace (~53 MB)
    char* p = (char*)d_ws;
    auto alloc = [&](size_t bytes) { char* q = p; p += (bytes + 255) & ~(size_t)255; return q; };
    float*  bufA  = (float*)alloc((size_t)N * HCH * 4);
    float*  bufB  = (float*)alloc((size_t)N * HCH * 4);
    float*  als   = (float*)alloc((size_t)N * 4);
    float*  ald   = (float*)alloc((size_t)N * 4);
    float*  mx    = (float*)alloc((size_t)N * 4);
    float*  dn    = (float*)alloc((size_t)N * 4);
    bf16_t* WT1   = (bf16_t*)alloc((size_t)64 * 128 * 2);
    bf16_t* WT2   = (bf16_t*)alloc((size_t)64 * 64 * 2);
    bf16_t* WT3   = (bf16_t*)alloc((size_t)64 * 64 * 2);
    float*  gmean = (float*)alloc((size_t)GNUM * HCH * 4);
    float*  gmax  = (float*)alloc((size_t)GNUM * HCH * 4);
    float*  gcnt  = (float*)alloc((size_t)GNUM * 4);

    const float NEGINF = -__builtin_huge_valf();

    wt_convert_kernel<<<(128 * HCH + 255) / 256, 256, 0, stream>>>(W[0], WT1, 128);
    wt_convert_kernel<<<(64 * HCH + 255) / 256, 256, 0, stream>>>(W[1], WT2, 64);
    wt_convert_kernel<<<(64 * HCH + 255) / 256, 256, 0, stream>>>(W[2], WT3, 64);

    const int gemmGrid = (N + 127) / 128;
    const int nodeGrid = (N + 255) / 256;
    const int nhGrid   = (N * HCH + 255) / 256;
    const int edgeGrid = (ET + 255) / 256;
    const int aggrGrid = (ET * 4 + 255) / 256;

    bf16_t* WTs[3] = {WT1, WT2, WT3};
    for (int l = 0; l < 3; ++l) {
        const float* Xin = (l == 0) ? x : bufB;
        if (l == 0)
            gat_gemm_wmma<128><<<gemmGrid, 256, 0, stream>>>(Xin, WTs[l], As[l], Ad[l],
                                                             bufA, als, ald, N);
        else
            gat_gemm_wmma<64><<<gemmGrid, 256, 0, stream>>>(Xin, WTs[l], As[l], Ad[l],
                                                            bufA, als, ald, N);
        fill_kernel<<<nodeGrid, 256, 0, stream>>>(mx, NEGINF, N);
        fill_kernel<<<nodeGrid, 256, 0, stream>>>(dn, 0.0f, N);
        fill_bias_kernel<<<nhGrid, 256, 0, stream>>>(bufB, B[l], N);   // bufB free after gemm read
        edge_max_kernel<<<edgeGrid, 256, 0, stream>>>(ei, E, ET, als, ald, mx);
        edge_expsum_kernel<<<edgeGrid, 256, 0, stream>>>(ei, E, ET, als, ald, mx, dn);
        edge_aggr_kernel<<<aggrGrid, 256, 0, stream>>>(ei, E, ET, als, ald, mx, dn, bufA, bufB);
        if (l < 2) relu_kernel<<<nhGrid, 256, 0, stream>>>(bufB, N);
    }

    fill_kernel<<<(GNUM * HCH + 255) / 256, 256, 0, stream>>>(gmean, 0.0f, GNUM * HCH);
    fill_kernel<<<(GNUM * HCH + 255) / 256, 256, 0, stream>>>(gmax, NEGINF, GNUM * HCH);
    fill_kernel<<<(GNUM + 255) / 256, 256, 0, stream>>>(gcnt, 0.0f, GNUM);
    pool_accum_kernel<<<nhGrid, 256, 0, stream>>>(bufB, batch, gmean, gmax, gcnt, N);
    head_kernel<<<(GNUM * C + 255) / 256, 256, 0, stream>>>(gmean, gmax, gcnt, lw, lb, out, C);
}
